// AxialAttention_6674379178342
// MI455X (gfx1250) — matrix operand order repaired and checked by reading, not compiled, not run on hardware
//
#include <hip/hip_runtime.h>
#include <hip/hip_bf16.h>

// ---------------------------------------------------------------------------
// AxialAttention for MI455X (gfx1250, wave32, WMMA bf16, TDM tile staging)
//   B=16, H=W=32, N=1024, C=1024, heads=16, head_dim=64, scale=0.125
// ---------------------------------------------------------------------------

typedef __attribute__((ext_vector_type(16))) __bf16 v16bf;
typedef __attribute__((ext_vector_type(8)))  __bf16 v8bf;
typedef __attribute__((ext_vector_type(8)))  float  v8f;
typedef __attribute__((ext_vector_type(4)))  unsigned int v4u;
typedef __attribute__((ext_vector_type(8)))  int    v8i;
typedef __attribute__((ext_vector_type(4)))  int    v4i;

#define C_DIM   1024
#define C3_DIM  3072
#define M_DIM   16384   // B * N tokens
#define HEADS   16
#define HDIM    64
#define SCALE   0.125f

#if defined(__has_builtin)
#if __has_builtin(__builtin_amdgcn_tensor_load_to_lds) && \
    __has_builtin(__builtin_amdgcn_s_wait_tensorcnt)
#define USE_TDM 1
#endif
#endif
#ifndef USE_TDM
#define USE_TDM 0
#endif

// ---- WMMA helper ----------------------------------------------------------
__device__ __forceinline__ v8f wmma_bf16(v16bf a, v16bf b, v8f c) {
  return __builtin_amdgcn_wmma_f32_16x16x32_bf16(
      /*neg_a=*/false, a, /*neg_b=*/false, b,
      /*c_mod=*/(short)0, c, /*reuse_a=*/false, /*reuse_b=*/false);
}

// A-matrix fragment (16x32, 16-bit): lane L<16 -> row L, K in {kk..+7, kk+16..+23}
//                                    lane L>=16 -> row L-16, K in {kk+8..+15, kk+24..+31}
__device__ __forceinline__ v16bf frag_a(const __bf16* base, int ld, int kk) {
  const int lane = threadIdx.x & 31;
  const int r    = lane & 15;
  const int ko   = kk + ((lane & 16) ? 8 : 0);
  const __bf16* p = base + r * ld + ko;
  v8bf lo = *(const v8bf*)p;
  v8bf hi = *(const v8bf*)(p + 16);
  return __builtin_shufflevector(lo, hi, 0,1,2,3,4,5,6,7,8,9,10,11,12,13,14,15);
}

// B-matrix fragment (32x16, 16-bit), fed from B^T storage (row = N, contiguous K):
//   lane L<16 -> col L, K in {kk..+7, kk+16..+23} ; lane L>=16 -> col L-16, K in {kk+8..+15, kk+24..+31}
__device__ __forceinline__ v16bf frag_b(const __bf16* base, int ld, int kk) {
  const int lane = threadIdx.x & 31;
  const int r    = lane & 15;
  const int ko   = kk + ((lane & 16) ? 8 : 0);
  const __bf16* p = base + r * ld + ko;
  v8bf lo = *(const v8bf*)p;
  v8bf hi = *(const v8bf*)(p + 16);
  return __builtin_shufflevector(lo, hi, 0,1,2,3,4,5,6,7,8,9,10,11,12,13,14,15);
}

#if USE_TDM
// ---- TDM: DMA a 128-row x 64-col bf16 tile into LDS with 8-elem row pad ---
// D# group0: count=1 | lds_addr | global_addr(57b) | type=2
// D# group1: data_size=2B, pad_enable, pad_interval=32 DWORD (code 4),
//            pad_amount=4 DWORD (code 3) -> LDS row stride 72 bf16.
//            tensor_dim0=1024, tensor_dim1=nrows, tile=64x128, stride0=1024.
// Toolchain here exposes the 6-arg builtin:
//   (uint32x4 g0, int32x8 g1, int32x4 g2, int32x4 g3, int32x8 extra, i32 cpol)
__device__ __forceinline__ void tdm_load_tile(const __bf16* gtile,
                                              unsigned lds_off,
                                              unsigned nrows) {
  const unsigned long long ga = (unsigned long long)gtile;
  v4u g0;
  g0[0] = 1u;                                            // count=1, user D#
  g0[1] = lds_off;                                       // LDS byte address
  g0[2] = (unsigned)(ga & 0xffffffffu);                  // global_addr[31:0]
  g0[3] = (unsigned)((ga >> 32) & 0x1ffffffu) | (2u << 30); // addr[56:32]|type=2
  v8i g1;
  g1[0] = (1 << 16) | (1 << 20) | (4 << 22) | (3 << 25); // dsize=2B,pad,int,amt
  g1[1] = (int)((C_DIM & 0xffffu) << 16);                // tensor_dim0 lo16
  g1[2] = (int)((C_DIM >> 16) | ((nrows & 0xffffu) << 16)); // dim0 hi|dim1 lo
  g1[3] = (int)((nrows >> 16) | (64u << 16));            // dim1 hi | tile_dim0
  g1[4] = (int)128;                                      // tile_dim1 (tile_dim2=0)
  g1[5] = (int)C_DIM;                                    // dim0_stride lo32
  g1[6] = 0;
  g1[7] = 0;
  const v4i z4 = {0, 0, 0, 0};                           // groups 2/3 unused (2D)
  const v8i z8 = {0, 0, 0, 0, 0, 0, 0, 0};
  __builtin_amdgcn_tensor_load_to_lds(g0, g1, z4, z4, z8, 0);
}
#endif

// ---- fp32 -> bf16 conversion ---------------------------------------------
__global__ __launch_bounds__(256) void cvt_bf16(const float* __restrict__ in,
                                                __bf16* __restrict__ out, int n) {
  int i = blockIdx.x * blockDim.x + threadIdx.x;
  if (i < n) out[i] = (__bf16)in[i];
}

// ---- QKV GEMM: C[16384,3072] = A[16384,1024] * W[3072,1024]^T -------------
// 256 threads (8 waves). 128x128 tile, K staged 64-wide, TDM double buffer.
__global__ __launch_bounds__(256) void qkv_gemm(const __bf16* __restrict__ A,
                                                const __bf16* __restrict__ Bw,
                                                __bf16* __restrict__ Cout) {
  // 2x(2x128x72) bf16 = 72 KB; epilogue tile aliases the A region (dead then).
  __shared__ __align__(16) __bf16 smem[36864];
  typedef __bf16 (*tile_t)[128][72];
  tile_t As = (tile_t)smem;                 // As[buf][row][k]
  tile_t Bs = (tile_t)(smem + 18432);       // Bs[buf][row][k]
  __bf16 (*Ct)[136] = (__bf16(*)[136])smem; // 128x136 bf16 = 34 KB (aliased)

  const int tid = threadIdx.x;
  const int m0  = blockIdx.y * 128;
  const int n0  = blockIdx.x * 128;
  const int wid = tid >> 5;
  const int mb  = (wid >> 1) * 32;
  const int nb  = (wid & 1) * 64;

#if USE_TDM
  auto issue_stage = [&](int s, int buf) {
    const int k0 = s * 64;
    if (wid == 0) {
      tdm_load_tile(A + (size_t)m0 * C_DIM + k0,
                    (unsigned)(size_t)&As[buf][0][0], M_DIM);
    } else if (wid == 1) {
      tdm_load_tile(Bw + (size_t)n0 * C_DIM + k0,
                    (unsigned)(size_t)&Bs[buf][0][0], C3_DIM);
    }
  };
#else
  auto issue_stage = [&](int s, int buf) {
    const int k0 = s * 64;
#pragma unroll
    for (int it = 0; it < 4; ++it) {
      const int i   = tid + it * 256;
      const int row = i >> 3;
      const int kc  = (i & 7) * 8;
      *(v8bf*)&As[buf][row][kc] = *(const v8bf*)&A [(size_t)(m0 + row) * C_DIM + k0 + kc];
      *(v8bf*)&Bs[buf][row][kc] = *(const v8bf*)&Bw[(size_t)(n0 + row) * C_DIM + k0 + kc];
    }
  };
#endif

  v8f acc[2][4] = {};

  issue_stage(0, 0);
#if USE_TDM
  __builtin_amdgcn_s_wait_tensorcnt(0);
#endif
  __syncthreads();

  int buf = 0;
  for (int s = 0; s < 16; ++s) {
    if (s + 1 < 16) issue_stage(s + 1, buf ^ 1);  // DMA next while computing
#pragma unroll
    for (int kk = 0; kk < 64; kk += 32) {
      v16bf a0 = frag_a(&As[buf][mb     ][0], 72, kk);
      v16bf a1 = frag_a(&As[buf][mb + 16][0], 72, kk);
      v16bf b0 = frag_b(&Bs[buf][nb     ][0], 72, kk);
      v16bf b1 = frag_b(&Bs[buf][nb + 16][0], 72, kk);
      v16bf b2 = frag_b(&Bs[buf][nb + 32][0], 72, kk);
      v16bf b3 = frag_b(&Bs[buf][nb + 48][0], 72, kk);
      acc[0][0] = wmma_bf16(a0, b0, acc[0][0]);
      acc[0][1] = wmma_bf16(a0, b1, acc[0][1]);
      acc[0][2] = wmma_bf16(a0, b2, acc[0][2]);
      acc[0][3] = wmma_bf16(a0, b3, acc[0][3]);
      acc[1][0] = wmma_bf16(a1, b0, acc[1][0]);
      acc[1][1] = wmma_bf16(a1, b1, acc[1][1]);
      acc[1][2] = wmma_bf16(a1, b2, acc[1][2]);
      acc[1][3] = wmma_bf16(a1, b3, acc[1][3]);
    }
#if USE_TDM
    __builtin_amdgcn_s_wait_tensorcnt(0);
#endif
    __syncthreads();
    buf ^= 1;
  }

  // Epilogue: stage tile in LDS, then coalesced b128 stores of bf16 rows.
  const int lane = tid & 31;
  const int col  = lane & 15;
  const int rsel = (lane & 16) ? 8 : 0;
#pragma unroll
  for (int i = 0; i < 2; ++i)
#pragma unroll
    for (int j = 0; j < 4; ++j)
#pragma unroll
      for (int r = 0; r < 8; ++r)
        Ct[mb + i * 16 + rsel + r][nb + j * 16 + col] = (__bf16)acc[i][j][r];
  __syncthreads();
#pragma unroll
  for (int it = 0; it < 8; ++it) {
    const int flat = tid + it * 256;   // 2048 chunks of 8 bf16
    const int row  = flat >> 4;
    const int c8   = (flat & 15) * 8;
    *(v8bf*)&Cout[(size_t)(m0 + row) * C3_DIM + n0 + c8] = *(const v8bf*)&Ct[row][c8];
  }
}

// ---- Axial attention over one (sequence, head). 128 threads (4 waves) -----
template <bool COL>
__global__ __launch_bounds__(128) void axial_attn(const __bf16* __restrict__ qkv,
                                                  const float* __restrict__ gamma,
                                                  const float* __restrict__ beta,
                                                  __bf16* __restrict__ outb,
                                                  float*  __restrict__ outf) {
  __shared__ __align__(16) __bf16 qs [32][72];   // LN(q)*scale
  __shared__ __align__(16) __bf16 kls[32][72];   // LN(k)
  __shared__ __align__(16) __bf16 vts[64][40];   // V transposed [dim][token]
  __shared__ __align__(16) float  ss [32][32];   // scores f32
  __shared__ __align__(16) __bf16 ps [32][40];   // softmax probs

  const int tid  = threadIdx.x;
  const int head = blockIdx.y;
  const int seq  = blockIdx.x;
  const int b    = seq >> 5;
  const int rw   = seq & 31;
  const int seq_base = b * 1024 + (COL ? rw : rw * 32);
  const int t_stride = COL ? 32 : 1;

  { // Phase 1: load + layernorm q,k; load v transposed
    const int tok  = tid >> 2;
    const int part = tid & 3;
    const int d0   = part * 16;
    const size_t m = (size_t)(seq_base + tok * t_stride);
    const __bf16* base = qkv + m * C3_DIM + head * HDIM + d0;

    float g[16], bt[16];
#pragma unroll
    for (int i = 0; i < 16; ++i) { g[i] = gamma[d0 + i]; bt[i] = beta[d0 + i]; }
    {
      float f[16], s1 = 0.f, s2 = 0.f;
#pragma unroll
      for (int i = 0; i < 16; ++i) { f[i] = (float)base[i]; s1 += f[i]; s2 += f[i] * f[i]; }
      s1 += __shfl_xor(s1, 1, 32); s1 += __shfl_xor(s1, 2, 32);
      s2 += __shfl_xor(s2, 1, 32); s2 += __shfl_xor(s2, 2, 32);
      const float mu  = s1 * (1.f / 64.f);
      const float var = s2 * (1.f / 64.f) - mu * mu;
      const float rs  = rsqrtf(var + 1e-5f);
#pragma unroll
      for (int i = 0; i < 16; ++i)
        qs[tok][d0 + i] = (__bf16)((((f[i] - mu) * rs) * g[i] + bt[i]) * SCALE);
    }
    {
      const __bf16* kb = base + C_DIM;
      float f[16], s1 = 0.f, s2 = 0.f;
#pragma unroll
      for (int i = 0; i < 16; ++i) { f[i] = (float)kb[i]; s1 += f[i]; s2 += f[i] * f[i]; }
      s1 += __shfl_xor(s1, 1, 32); s1 += __shfl_xor(s1, 2, 32);
      s2 += __shfl_xor(s2, 1, 32); s2 += __shfl_xor(s2, 2, 32);
      const float mu  = s1 * (1.f / 64.f);
      const float var = s2 * (1.f / 64.f) - mu * mu;
      const float rs  = rsqrtf(var + 1e-5f);
#pragma unroll
      for (int i = 0; i < 16; ++i)
        kls[tok][d0 + i] = (__bf16)(((f[i] - mu) * rs) * g[i] + bt[i]);
    }
    {
      const __bf16* vb = base + 2 * C_DIM;
#pragma unroll
      for (int i = 0; i < 16; ++i) vts[d0 + i][tok] = vb[i];
    }
  }
  __syncthreads();

  const int wid  = tid >> 5;
  const int lane = tid & 31;
  const int col  = lane & 15;
  const int rsel = (lane & 16) ? 8 : 0;

  { // Phase 2: S = (q*scale) @ k^T ; one 16x16 tile per wave
    const int ti = (wid >> 1) * 16;
    const int tj = (wid & 1) * 16;
    v8f sacc = {};
#pragma unroll
    for (int kk = 0; kk < 64; kk += 32) {
      v16bf aq = frag_a(&qs [ti][0], 72, kk);
      v16bf bk = frag_b(&kls[tj][0], 72, kk);
      sacc = wmma_bf16(aq, bk, sacc);
    }
#pragma unroll
    for (int r = 0; r < 8; ++r) ss[ti + rsel + r][tj + col] = sacc[r];
  }
  __syncthreads();

  { // Phase 3: softmax (4 lanes x 8 cols per row)
    const int row = tid >> 2;
    const int c0  = (tid & 3) * 8;
    float v[8], mx = -3.0e38f;
#pragma unroll
    for (int i = 0; i < 8; ++i) { v[i] = ss[row][c0 + i]; mx = fmaxf(mx, v[i]); }
    mx = fmaxf(mx, __shfl_xor(mx, 1, 32));
    mx = fmaxf(mx, __shfl_xor(mx, 2, 32));
    float sum = 0.f;
#pragma unroll
    for (int i = 0; i < 8; ++i) { v[i] = __expf(v[i] - mx); sum += v[i]; }
    sum += __shfl_xor(sum, 1, 32);
    sum += __shfl_xor(sum, 2, 32);
    const float inv = 1.f / sum;
#pragma unroll
    for (int i = 0; i < 8; ++i) ps[row][c0 + i] = (__bf16)(v[i] * inv);
  }
  __syncthreads();

  { // Phase 4: O = P @ V ; each wave 1 M-tile x 2 N-tiles
    const int tm  = (wid >> 1) * 16;
    const int jt0 = (wid & 1) * 2;
    v16bf ap = frag_a(&ps[tm][0], 40, 0);
#pragma unroll
    for (int j = 0; j < 2; ++j) {
      const int nt = (jt0 + j) * 16;
      v16bf bv = frag_b(&vts[nt][0], 40, 0);
      v8f o = {};
      o = wmma_bf16(ap, bv, o);
      const int n = head * HDIM + nt + col;
#pragma unroll
      for (int r = 0; r < 8; ++r) {
        const int t = tm + rsel + r;
        const size_t idx = (size_t)(seq_base + t * t_stride) * C_DIM + n;
        if (COL) outf[idx] = o[r];
        else     outb[idx] = (__bf16)o[r];
      }
    }
  }
}

// ---------------------------------------------------------------------------
extern "C" void kernel_launch(void* const* d_in, const int* in_sizes, int n_in,
                              void* d_out, int out_size, void* d_ws, size_t ws_size,
                              hipStream_t stream) {
  (void)in_sizes; (void)n_in; (void)out_size; (void)ws_size;

  const float* x  = (const float*)d_in[0];
  const float* Wr = (const float*)d_in[1];
  const float* Wc = (const float*)d_in[2];
  const float* gr = (const float*)d_in[3];
  const float* br = (const float*)d_in[4];
  const float* gc = (const float*)d_in[5];
  const float* bc = (const float*)d_in[6];
  float* out      = (float*)d_out;

  char* ws = (char*)d_ws;
  __bf16* xb   = (__bf16*)(ws);                          // 32 MB
  __bf16* mid  = (__bf16*)(ws + (size_t)32 * 1048576);   // 32 MB
  __bf16* wrb  = (__bf16*)(ws + (size_t)64 * 1048576);   //  6 MB
  __bf16* wcb  = (__bf16*)(ws + (size_t)72 * 1048576);   //  6 MB
  __bf16* qkvb = (__bf16*)(ws + (size_t)80 * 1048576);   // 96 MB

  const int nx = M_DIM * C_DIM;
  const int nw = C3_DIM * C_DIM;
  cvt_bf16<<<(nx + 255) / 256, 256, 0, stream>>>(x,  xb,  nx);
  cvt_bf16<<<(nw + 255) / 256, 256, 0, stream>>>(Wr, wrb, nw);
  cvt_bf16<<<(nw + 255) / 256, 256, 0, stream>>>(Wc, wcb, nw);

  const dim3 ggrid(C3_DIM / 128, M_DIM / 128);   // 24 x 128
  const dim3 agrid(512, HEADS);                  // sequences x heads

  qkv_gemm<<<ggrid, 256, 0, stream>>>(xb, wrb, qkvb);
  axial_attn<false><<<agrid, 128, 0, stream>>>(qkvb, gr, br, mid, nullptr);

  qkv_gemm<<<ggrid, 256, 0, stream>>>(mid, wcb, qkvb);
  axial_attn<true><<<agrid, 128, 0, stream>>>(qkvb, gc, bc, nullptr, out);
}
